// ConvLSTM2D_a_18494129176662
// MI455X (gfx1250) — matrix operand order repaired and checked by reading, not compiled, not run on hardware
//
#include <hip/hip_runtime.h>

// ---------------------------------------------------------------------------
// ConvLSTM2D (B=16,T=32,112x112x3 -> 28x28x32) + MLP(1024,256,6) + softmax
// CDNA5 / gfx1250, wave32, WMMA f16 (convs) + WMMA f32 16x16x4 (dense layers)
// ---------------------------------------------------------------------------

typedef __attribute__((ext_vector_type(16))) _Float16 v16h;
typedef __attribute__((ext_vector_type(8)))  float    v8f;
typedef __attribute__((ext_vector_type(2)))  float    v2f;

#define NB    16          // batch
#define NT    32          // timesteps
#define P     784         // 28*28 output pixels
#define NF    32          // LSTM filters
#define NCH   128         // 4*NF gate channels
#define KSREC 6           // 192 / 32 k-steps for recurrent conv
#define MIN_TOT (NT*NB*P) // 401408 rows of input-conv GEMM
#define MSTEP  (NB*P)     // 12544 rows per recurrent step

__device__ __forceinline__ float hsig(float z) {
    return fminf(fmaxf(0.2f * z + 0.5f, 0.0f), 1.0f);
}

// ---------------------------------------------------------------------------
// Pack conv weights (fp32 row-major [K][128]) into WMMA B-fragment layout
// (16-bit B 32x16, contiguous k order used here, matching A: lane = n%16 + 16*(k_local/16), elem = k_local%16).
// Input conv: K=18 padded to 32 (1 k-step). Rec conv: K=192 (6 k-steps).
// frag index = ((nt*KS + ks)*32 + lane)*16 + e
// ---------------------------------------------------------------------------
__global__ void pack_w_kernel(const float* __restrict__ kin,
                              const float* __restrict__ krec,
                              _Float16* __restrict__ wfin,
                              _Float16* __restrict__ wfrec) {
    int i = blockIdx.x * blockDim.x + threadIdx.x;
    if (i < 8 * 512) {                       // input-conv frags: 8 ntiles, 1 kstep
        int e = i & 15, lane = (i >> 4) & 31, nt = i >> 9;
        int kl = e + ((lane >> 4) << 4);     // k within k-step (0..31)
        int n  = nt * 16 + (lane & 15);
        float v = (kl < 18) ? kin[kl * NCH + n] : 0.0f;
        wfin[i] = (_Float16)v;
    }
    int j = i - 8 * 512;
    if (j >= 0 && j < 8 * KSREC * 512) {     // rec-conv frags
        int e = j & 15, lane = (j >> 4) & 31, rest = j >> 9;
        int ks = rest % KSREC, nt = rest / KSREC;
        int kl = ks * 32 + e + ((lane >> 4) << 4);   // 0..191
        int n  = nt * 16 + (lane & 15);
        wfrec[j] = (_Float16)krec[kl * NCH + n];
    }
}

// Zero h0 (f16) and cell state (f32)
__global__ void init_state_kernel(_Float16* __restrict__ h0, float* __restrict__ c) {
    int i = blockIdx.x * blockDim.x + threadIdx.x;
    if (i < MSTEP * NF) { h0[i] = (_Float16)0.0f; c[i] = 0.0f; }
}

// ---------------------------------------------------------------------------
// Input conv for ALL timesteps as one big GEMM:
//   A[m, k<=32] im2col of x (m = (t*16+b)*784 + p), B = wfin, D -> zin (f16)
// Block: 128 rows (8 m-tiles), 8 waves: wave w owns m-tile w, loops 8 n-tiles.
// ---------------------------------------------------------------------------
__global__ __launch_bounds__(256) void inconv_kernel(
    const float* __restrict__ x, const _Float16* __restrict__ wfin,
    _Float16* __restrict__ zin) {
    __shared__ __align__(32) _Float16 lA[8 * 32 * 16];   // 8 m-tiles, frag layout
    const int tid = threadIdx.x;
    const int m0  = blockIdx.x * 128;

    for (int idx = tid; idx < 128 * 32; idx += 256) {
        int ml = idx >> 5, kk = idx & 31;
        int mg = m0 + ml;
        int p  = mg % P;  int tb = mg / P;
        int b  = tb & 15; int t  = tb >> 4;
        float v = 0.0f;
        if (kk < 18) {                        // k = (dy*3+dx)*3 + ci
            int ci = kk % 3, q = kk / 3;
            int dx = q % 3,  dy = q / 3;
            int iy = (p / 28) * 4 + dy;       // stride 4, VALID: always in-bounds
            int ix = (p % 28) * 4 + dx;
            v = x[(size_t)(((b * NT + t) * 112 + iy) * 112 + ix) * 3 + ci];
        }
        int mt = ml >> 4, row = ml & 15;      // A-frag layout (16-bit 16x32)
        int lane = row + ((kk >> 4) << 4);
        int e    = kk & 15;
        lA[(mt * 32 + lane) * 16 + e] = (_Float16)v;
    }
    __syncthreads();

    int w = tid >> 5, lane = tid & 31;
    v16h a = *reinterpret_cast<const v16h*>(&lA[(w * 32 + lane) * 16]);
    int mbase = m0 + w * 16 + ((lane >> 4) << 3);
    for (int nt = 0; nt < 8; ++nt) {
        v16h bw = *reinterpret_cast<const v16h*>(&wfin[(nt * 32 + lane) * 16]);
        v8f cacc = {};
        cacc = __builtin_amdgcn_wmma_f32_16x16x32_f16(false, a, false, bw,
                                                      (short)0, cacc, false, false);
        int ch = nt * 16 + (lane & 15);
        #pragma unroll
        for (int r = 0; r < 8; ++r)
            zin[(size_t)(mbase + r) * NCH + ch] = (_Float16)cacc[r];
    }
}

// ---------------------------------------------------------------------------
// One LSTM timestep: z = zin[t] + conv_SAME(h_prev, rec_kernel) + bias; gates.
// Block: 64 pixels (4 m-tiles). 8 waves = 8 n-tiles, K=192 (6 k-steps).
// z tile staged in LDS so gate math sees all 128 channels per pixel.
// ---------------------------------------------------------------------------
__global__ __launch_bounds__(256) void lstm_step_kernel(
    const _Float16* __restrict__ hin, _Float16* __restrict__ hout,
    float* __restrict__ cst, const _Float16* __restrict__ zin,
    const _Float16* __restrict__ wfrec, const float* __restrict__ bias, int t) {
    __shared__ __align__(32) _Float16 lA[4 * KSREC * 32 * 16];  // 24 KB
    __shared__ float zt[64 * NCH];                               // 32 KB
    const int tid = threadIdx.x;
    const int m0  = blockIdx.x * 64;

    // im2col gather of h_prev (SAME pad: top 0 / bottom 1, left 1 / right 1)
    for (int idx = tid; idx < 64 * 192; idx += 256) {
        int ml = idx / 192; int k = idx - ml * 192;
        int mg = m0 + ml;  int b = mg / P;  int p = mg - b * P;
        int y = p / 28, xq = p - y * 28;
        int s = k >> 5, ci = k & 31;          // k = (dy*3+dx)*32 + ci
        int dy = (s >= 3) ? 1 : 0;
        int dx = s - dy * 3;
        int iy = y + dy, ix = xq + dx - 1;
        float v = 0.0f;
        if (iy < 28 && (unsigned)ix < 28u)
            v = (float)hin[((size_t)b * P + iy * 28 + ix) * NF + ci];
        int mt = ml >> 4, row = ml & 15;
        int lane = row + ((ci >> 4) << 4);
        int e    = ci & 15;
        lA[((mt * KSREC + s) * 32 + lane) * 16 + e] = (_Float16)v;
    }
    __syncthreads();

    {
        int w = tid >> 5, lane = tid & 31;    // w = n-tile
        int ch = w * 16 + (lane & 15);
        #pragma unroll
        for (int mt = 0; mt < 4; ++mt) {
            v8f cacc = {};
            #pragma unroll
            for (int ks = 0; ks < KSREC; ++ks) {
                v16h a  = *reinterpret_cast<const v16h*>(
                              &lA[((mt * KSREC + ks) * 32 + lane) * 16]);
                v16h bw = *reinterpret_cast<const v16h*>(
                              &wfrec[((w * KSREC + ks) * 32 + lane) * 16]);
                cacc = __builtin_amdgcn_wmma_f32_16x16x32_f16(
                           false, a, false, bw, (short)0, cacc, false, false);
            }
            int mrow = mt * 16 + ((lane >> 4) << 3);
            #pragma unroll
            for (int r = 0; r < 8; ++r) zt[(mrow + r) * NCH + ch] = cacc[r];
        }
    }
    __syncthreads();

    // gate elementwise: 64 pixels x 32 features
    for (int idx = tid; idx < 64 * NF; idx += 256) {
        int ml = idx >> 5, ff = idx & 31;
        int mg = m0 + ml; int b = mg / P; int p = mg - b * P;
        size_t zb = ((size_t)((t * NB + b) * P + p)) * NCH;
        float zi = zt[ml * NCH + ff]      + (float)zin[zb + ff]      + bias[ff];
        float zf = zt[ml * NCH + 32 + ff] + (float)zin[zb + 32 + ff] + bias[32 + ff];
        float zc = zt[ml * NCH + 64 + ff] + (float)zin[zb + 64 + ff] + bias[64 + ff];
        float zo = zt[ml * NCH + 96 + ff] + (float)zin[zb + 96 + ff] + bias[96 + ff];
        int sidx = mg * NF + ff;
        float cn = hsig(zf) * cst[sidx] + hsig(zi) * tanhf(zc);
        cst[sidx]  = cn;
        hout[sidx] = (_Float16)(hsig(zo) * tanhf(cn));
    }
}

// ---------------------------------------------------------------------------
// fc1: [16,25088] @ [25088,1024] via fp32 WMMA 16x16x4 (streams w1 as fp32).
// Grid (64 ntiles, 2); 8 waves/block -> 16 deterministic K-partials of 1568.
// ---------------------------------------------------------------------------
__global__ __launch_bounds__(256) void fc1_kernel(
    const _Float16* __restrict__ h, const float* __restrict__ w1,
    float* __restrict__ part) {
    int w = threadIdx.x >> 5, lane = threadIdx.x & 31;
    int nt = blockIdx.x;
    int kc = blockIdx.y * 8 + w;              // 0..15
    int n  = nt * 16 + (lane & 15);
    int m  = lane & 15;                       // batch row for A
    int kh = (lane >> 4) << 1;                // 0 or 2
    int kbase = kc * 1568;
    v8f cacc = {};
    for (int kk = 0; kk < 1568; kk += 4) {
        int k0 = kbase + kk + kh;
        v2f a, bv;
        a.x  = (float)h[(size_t)m * 25088 + k0];
        a.y  = (float)h[(size_t)m * 25088 + k0 + 1];
        bv.x = w1[(size_t)k0 * 1024 + n];
        bv.y = w1[(size_t)(k0 + 1) * 1024 + n];
        cacc = __builtin_amdgcn_wmma_f32_16x16x4_f32(false, a, false, bv,
                                                     (short)0, cacc, false, false);
    }
    int mrow = (lane >> 4) << 3;
    #pragma unroll
    for (int r = 0; r < 8; ++r)
        part[((size_t)kc * 16 + mrow + r) * 1024 + n] = cacc[r];
}

__global__ void fc1_reduce_kernel(const float* __restrict__ part,
                                  const float* __restrict__ b1,
                                  float* __restrict__ zws) {
    int i = blockIdx.x * blockDim.x + threadIdx.x;
    if (i < 16 * 1024) {
        float s = b1[i & 1023];
        #pragma unroll
        for (int kc = 0; kc < 16; ++kc) s += part[(size_t)kc * 16384 + i];
        zws[i] = s;
    }
}

// fc2: [16,1024] @ [1024,256] + b2, fp32 WMMA. 16 blocks x 1 wave.
__global__ void fc2_kernel(const float* __restrict__ zws,
                           const float* __restrict__ w2,
                           const float* __restrict__ b2,
                           float* __restrict__ out2) {
    int lane = threadIdx.x & 31;
    int nt = blockIdx.x;
    int n  = nt * 16 + (lane & 15);
    int m  = lane & 15;
    int kh = (lane >> 4) << 1;
    v8f cacc = {};
    for (int k = 0; k < 1024; k += 4) {
        v2f a, bv;
        a.x  = zws[m * 1024 + k + kh];
        a.y  = zws[m * 1024 + k + kh + 1];
        bv.x = w2[(k + kh) * 256 + n];
        bv.y = w2[(k + kh + 1) * 256 + n];
        cacc = __builtin_amdgcn_wmma_f32_16x16x4_f32(false, a, false, bv,
                                                     (short)0, cacc, false, false);
    }
    int mrow = (lane >> 4) << 3;
    #pragma unroll
    for (int r = 0; r < 8; ++r)
        out2[(mrow + r) * 256 + n] = cacc[r] + b2[n];
}

// fc3 + softmax: one lane per batch row (tiny)
__global__ void fc3_softmax_kernel(const float* __restrict__ f2,
                                   const float* __restrict__ w3,
                                   const float* __restrict__ b3,
                                   float* __restrict__ out) {
    int b = threadIdx.x;
    if (b < NB) {
        float lg[6];
        #pragma unroll
        for (int j = 0; j < 6; ++j) {
            float s = b3[j];
            for (int k = 0; k < 256; ++k) s += f2[b * 256 + k] * w3[k * 6 + j];
            lg[j] = s;
        }
        float mx = lg[0];
        #pragma unroll
        for (int j = 1; j < 6; ++j) mx = fmaxf(mx, lg[j]);
        float sum = 0.0f;
        #pragma unroll
        for (int j = 0; j < 6; ++j) { lg[j] = expf(lg[j] - mx); sum += lg[j]; }
        float inv = 1.0f / sum;
        #pragma unroll
        for (int j = 0; j < 6; ++j) out[b * 6 + j] = lg[j] * inv;
    }
}

// ---------------------------------------------------------------------------
extern "C" void kernel_launch(void* const* d_in, const int* in_sizes, int n_in,
                              void* d_out, int out_size, void* d_ws, size_t ws_size,
                              hipStream_t stream) {
    const float* x    = (const float*)d_in[0];
    const float* kin  = (const float*)d_in[1];
    const float* krec = (const float*)d_in[2];
    const float* bias = (const float*)d_in[3];
    const float* w1   = (const float*)d_in[4];
    const float* b1   = (const float*)d_in[5];
    const float* w2   = (const float*)d_in[6];
    const float* b2   = (const float*)d_in[7];
    const float* w3   = (const float*)d_in[8];
    const float* b3   = (const float*)d_in[9];
    float* out = (float*)d_out;

    char* ws = (char*)d_ws;
    _Float16* zin   = (_Float16*)ws;  ws += (size_t)MIN_TOT * NCH * 2;   // 102.8 MB
    _Float16* hA    = (_Float16*)ws;  ws += (size_t)MSTEP * NF * 2;      // h ping
    _Float16* hB    = (_Float16*)ws;  ws += (size_t)MSTEP * NF * 2;      // h pong
    float*    cst   = (float*)ws;     ws += (size_t)MSTEP * NF * 4;      // cell state
    _Float16* wfin  = (_Float16*)ws;  ws += 8 * 512 * 2;                 // in-conv frags
    _Float16* wfrec = (_Float16*)ws;  ws += 8 * KSREC * 512 * 2;         // rec-conv frags
    float*    part  = (float*)ws;     ws += (size_t)16 * 16 * 1024 * 4;  // fc1 K-partials
    float*    zws   = (float*)ws;     ws += 16 * 1024 * 4;               // fc1 out
    float*    f2o   = (float*)ws;     ws += 16 * 256 * 4;                // fc2 out

    pack_w_kernel<<<112, 256, 0, stream>>>(kin, krec, wfin, wfrec);
    init_state_kernel<<<(MSTEP * NF + 255) / 256, 256, 0, stream>>>(hA, cst);

    inconv_kernel<<<MIN_TOT / 128, 256, 0, stream>>>(x, wfin, zin);

    for (int t = 0; t < NT; ++t) {
        const _Float16* hin = (t & 1) ? hB : hA;
        _Float16*       hout = (t & 1) ? hA : hB;
        lstm_step_kernel<<<MSTEP / 64, 256, 0, stream>>>(hin, hout, cst, zin,
                                                         wfrec, bias, t);
    }
    // after 32 steps (even count), final h lives in hA

    fc1_kernel<<<dim3(64, 2), 256, 0, stream>>>(hA, w1, part);
    fc1_reduce_kernel<<<64, 256, 0, stream>>>(part, b1, zws);
    fc2_kernel<<<16, 32, 0, stream>>>(zws, w2, b2, f2o);
    fc3_softmax_kernel<<<1, 32, 0, stream>>>(f2o, w3, b3, out);
}
